// GATLayer_12876311954175
// MI455X (gfx1250) — matrix operand order repaired and checked by reading, not compiled, not run on hardware
//
#include <hip/hip_runtime.h>
#include <hip/hip_bf16.h>
#include <stdint.h>

typedef __bf16 bf16_t;
typedef __attribute__((ext_vector_type(16))) __bf16 v16bf;
typedef __attribute__((ext_vector_type(8)))  __bf16 v8bf;
typedef __attribute__((ext_vector_type(8)))  float  v8f;
typedef __attribute__((ext_vector_type(4)))  float  v4f;
typedef __attribute__((ext_vector_type(4)))  unsigned v4u;

#define NN      8192
#define FIN     512
#define FOUT    64
#define MWORDS  (NN / 32)          /* 256 mask words per row  */
#define KSPLIT  4
#define KSPS    (MWORDS / KSPLIT)  /* 64 K-steps per split    */

#if __has_builtin(__builtin_amdgcn_exp2f)
#define EXP2F(x) __builtin_amdgcn_exp2f(x)
#else
#define EXP2F(x) exp2f(x)
#endif

#define LOG2E 1.4426950408889634f

// ---------------------------------------------------------------------------
// 1) adj (int32, 256 MB) -> bitmask (8 MB).  One wave32 ballot per 32 entries.
// ---------------------------------------------------------------------------
__global__ void pack_kernel(const int* __restrict__ adj, unsigned* __restrict__ mask) {
    unsigned gid = blockIdx.x * 256u + threadIdx.x;
    int v = adj[gid];
    unsigned long long b = __ballot(v > 0);
    if ((threadIdx.x & 31u) == 0u) mask[gid >> 5] = (unsigned)b;
}

// ---------------------------------------------------------------------------
// 2) Wh = h @ W  (f32) and Wh^T in bf16 ([n][j] so WMMA B-tiles are
//    two 16 B runs per lane).
// ---------------------------------------------------------------------------
__global__ void wh_kernel(const float* __restrict__ h, const float* __restrict__ W,
                          float* __restrict__ Wh, bf16_t* __restrict__ WhT) {
    int j = blockIdx.x;         // node
    int n = threadIdx.x;        // out feature 0..63
    const float* hr = h + (size_t)j * FIN;
    float acc = 0.f;
#pragma unroll 8
    for (int k = 0; k < FIN; ++k) acc = fmaf(hr[k], W[k * FOUT + n], acc);
    Wh[(size_t)j * FOUT + n] = acc;
    WhT[(size_t)n * NN + j] = (bf16_t)acc;
}

// ---------------------------------------------------------------------------
// 3) f_src = Wh @ a[:64], f_dst = Wh @ a[64:]   (both pre-scaled by log2(e))
// ---------------------------------------------------------------------------
__global__ void feat_kernel(const float* __restrict__ Wh, const float* __restrict__ a,
                            float* __restrict__ fsrc, float* __restrict__ fdst) {
    int j = blockIdx.x * 256 + threadIdx.x;
    const float* wr = Wh + (size_t)j * FOUT;
    float s = 0.f, d = 0.f;
#pragma unroll
    for (int n = 0; n < FOUT; ++n) {
        float v = wr[n];
        s = fmaf(v, a[n], s);
        d = fmaf(v, a[FOUT + n], d);
    }
    fsrc[j] = s * LOG2E;            // log2-domain: exp(x) == exp2(log2e*x),
    fdst[j] = d * LOG2E;            // leakyrelu commutes with positive scaling
}

// ---------------------------------------------------------------------------
// 4) global max of (scaled) f_dst — safe softmax shift
// ---------------------------------------------------------------------------
__global__ void dmax_kernel(const float* __restrict__ fdst, float* __restrict__ dmax) {
    __shared__ float red[256];
    float m = -3.4e38f;
    for (int j = threadIdx.x; j < NN; j += 256) m = fmaxf(m, fdst[j]);
    red[threadIdx.x] = m;
    __syncthreads();
    for (int s = 128; s > 0; s >>= 1) {
        if ((int)threadIdx.x < s) red[threadIdx.x] = fmaxf(red[threadIdx.x], red[threadIdx.x + s]);
        __syncthreads();
    }
    if (threadIdx.x == 0) *dmax = red[0];
}

// ---------------------------------------------------------------------------
// 5) Fused masked-softmax-matmul.  One wave owns a 16-row tile for one K
//    split.  B-tiles AND mask-word groups are double-buffered so no load is
//    waited on sooner than a full step (B) / four steps (mask) after issue.
// ---------------------------------------------------------------------------
__global__ __launch_bounds__(128)
void attn_kernel(const unsigned* __restrict__ mask, const bf16_t* __restrict__ WhT,
                 const float* __restrict__ fsrc, const float* __restrict__ fdst,
                 const float* __restrict__ dmaxp,
                 float* __restrict__ accW, float* __restrict__ rsum) {
    __shared__ float dsl [KSPS * 32];             // f_dst slice (log2-domain)
    __shared__ float dsl5[KSPS * 32];             // 0.2 * same (leakyrelu arm)

    const int split = blockIdx.y;
    const int kw0   = split * KSPS;               // first mask word of slice

    {   // cooperative vectorized LDS fill
        const v4f* s4 = (const v4f*)(fdst + kw0 * 32);
        v4f* d4 = (v4f*)dsl;
        v4f* d5 = (v4f*)dsl5;
        for (int i = threadIdx.x; i < (KSPS * 32) / 4; i += 128) {
            v4f v = s4[i];
            d4[i] = v;
            d5[i] = v * 0.2f;
        }
    }
    __syncthreads();

    const int wave = threadIdx.x >> 5;
    const int lane = threadIdx.x & 31;
    const int half = lane >> 4;                   // wave32 half
    const int m    = lane & 15;                   // row in tile / col in N-tile
    const int rowbase = (blockIdx.x * 4 + wave) * 16;
    const int row  = rowbase + m;

    const float dmax = *dmaxp;                    // log2-scaled
    const float s2 = fsrc[row];                   // log2-scaled
    float M2 = s2 + dmax;
    M2 = fmaxf(M2, 0.2f * M2);                    // leakyrelu == max(x, 0.2x)
    const float sA = s2 - M2;                     // arg = max(sA+d, sB+0.2d)
    const float sB = 0.2f * s2 - M2;
    const int   shamt = half << 3;                // pre-shift for literal bit tests

    v8f acc[4] = {};                              // 16x64 f32 accumulators
    float rowsum = 0.f;

    const unsigned* mrow = mask + (size_t)row * MWORDS + kw0;   // 256B aligned
    const bf16_t*   brow = WhT + (size_t)m * NN + (size_t)kw0 * 32 + half * 8;

    // double buffers: B [parity][tile], mask groups [group parity]
    v8bf blo[2][4], bhi[2][4];
    v4u  wbuf[2] = {};

#define LOAD_B(ks_, par_)                                                     \
    {                                                                         \
        const bf16_t* bp_ = brow + (ks_) * 32;                                \
        _Pragma("unroll")                                                     \
        for (int t = 0; t < 4; ++t) {                                         \
            blo[par_][t] = *(const v8bf*)(bp_ + (size_t)t * 16 * NN);         \
            bhi[par_][t] = *(const v8bf*)(bp_ + (size_t)t * 16 * NN + 16);    \
        }                                                                     \
    }

    LOAD_B(0, 0)                                   // prologue
    wbuf[0] = *(const v4u*)(mrow);

#pragma unroll 8                                   // makes ks&1 and (ks>>2)&1 literal
    for (int ks = 0; ks < KSPS; ++ks) {
        const int cur = ks & 1, nxt = cur ^ 1;
        const int g   = (ks >> 2) & 1;

        // ---- 1) prefetch next step's B-tiles (full body of latency cover) ----
        if (ks + 1 < KSPS) LOAD_B(ks + 1, nxt)

        // ---- 2) prefetch NEXT mask group; consume the one loaded 4 steps ago ----
        if ((ks & 3) == 0 && ks + 4 < KSPS)
            wbuf[g ^ 1] = *(const v4u*)(mrow + ks + 4);
        const unsigned wsh = wbuf[g][ks & 3] >> shamt;  // bit e -> (e>>3)*16+(e&7)

        // ---- 3) lane's 16 d / 0.2d values (8x ds_load_b128) ----
        const int kl = ks * 32;
        const float* dp  = dsl  + kl + shamt;      // half*8 floats
        const float* dp5 = dsl5 + kl + shamt;
        const v4f dA = *(const v4f*)(dp);      const v4f eA = *(const v4f*)(dp5);
        const v4f dB = *(const v4f*)(dp + 4);  const v4f eB = *(const v4f*)(dp5 + 4);
        const v4f dC = *(const v4f*)(dp + 16); const v4f eC = *(const v4f*)(dp5 + 16);
        const v4f dD = *(const v4f*)(dp + 20); const v4f eD = *(const v4f*)(dp5 + 20);
        float dv[16], dv5[16];
#pragma unroll
        for (int i = 0; i < 4; ++i) {
            dv [i] = dA[i]; dv [4 + i] = dB[i]; dv [8 + i] = dC[i]; dv [12 + i] = dD[i];
            dv5[i] = eA[i]; dv5[4 + i] = eB[i]; dv5[8 + i] = eC[i]; dv5[12 + i] = eD[i];
        }

        // ---- 4) branchless A-tile; bit positions are literals ----
        v16bf A;
        float rs0 = 0.f, rs1 = 0.f;
#pragma unroll
        for (int e = 0; e < 16; ++e) {
            const int bit = ((e >> 3) << 4) + (e & 7);          // literal
            const float arg = fmaxf(sA + dv[e], sB + dv5[e]);   // leakyrelu-M2
            const float ev = EXP2F(arg);                        // v_exp_f32
            const float wgt = (wsh & (1u << bit)) ? ev : 0.f;   // and-lit+cndmask
            if (e & 1) rs1 += wgt; else rs0 += wgt;
            A[e] = (bf16_t)wgt;
        }
        rowsum += rs0 + rs1;

        // ---- 5) 4 WMMAs from the buffer loaded LAST step ----
#pragma unroll
        for (int t = 0; t < 4; ++t) {
            v16bf B;
#pragma unroll
            for (int e = 0; e < 8; ++e) { B[e] = blo[cur][t][e]; B[8 + e] = bhi[cur][t][e]; }
            acc[t] = __builtin_amdgcn_wmma_f32_16x16x32_bf16(
                false, A, false, B, (short)0, acc[t], false, false);
        }
    }
#undef LOAD_B

    // combine half-wave partial row sums (lanes m and m+16 -> full sum)
    rowsum += __shfl_xor(rowsum, 16, 32);
    if (lane < 16) rsum[(size_t)split * NN + rowbase + m] = rowsum;

    // C/D layout: VGPR r, lanes 0-15 -> M=r ; lanes 16-31 -> M=r+8 ; N = lane&15
    float* accS = accW + (size_t)split * NN * FOUT;
#pragma unroll
    for (int r = 0; r < 8; ++r) {
        const int orow = rowbase + r + (half << 3);
#pragma unroll
        for (int t = 0; t < 4; ++t)
            accS[(size_t)orow * FOUT + t * 16 + m] = acc[t][r];
    }
}

// ---------------------------------------------------------------------------
// 6) reduce K-split slices, normalize by row sum, ELU, store f32 output
// ---------------------------------------------------------------------------
__global__ void finalize_kernel(const float* __restrict__ accW, const float* __restrict__ rsum,
                                float* __restrict__ out) {
    int idx = blockIdx.x * 256 + threadIdx.x;     // 0 .. NN*FOUT-1
    int row = idx >> 6;
    float a = 0.f, ss = 0.f;
#pragma unroll
    for (int s = 0; s < KSPLIT; ++s) a += accW[(size_t)s * NN * FOUT + idx];
#pragma unroll
    for (int s = 0; s < KSPLIT; ++s) ss += rsum[(size_t)s * NN + row];
    float v = (ss > 0.f) ? a / ss : 0.f;
    v = (v > 0.f) ? v : (EXP2F(v * LOG2E) - 1.f); // ELU(alpha=1)
    out[idx] = v;
}

// ---------------------------------------------------------------------------
extern "C" void kernel_launch(void* const* d_in, const int* in_sizes, int n_in,
                              void* d_out, int out_size, void* d_ws, size_t ws_size,
                              hipStream_t stream) {
    const float* h   = (const float*)d_in[0];
    const int*   adj = (const int*)d_in[1];
    const float* W   = (const float*)d_in[2];
    const float* a   = (const float*)d_in[3];
    float* out = (float*)d_out;

    char* ws = (char*)d_ws;
    const size_t MB = 1024ull * 1024ull;
    unsigned* mask = (unsigned*)(ws);                     // 8 MB
    float*    Wh   = (float*)   (ws +  8 * MB);           // 2 MB
    bf16_t*   WhT  = (bf16_t*)  (ws + 10 * MB);           // 1 MB
    float*    fsrc = (float*)   (ws + 11 * MB);           // 32 KB
    float*    fdst = (float*)   (ws + 11 * MB + 32768);   // 32 KB
    float*    dmax = (float*)   (ws + 11 * MB + 65536);   // 4 B
    float*    accW = (float*)   (ws + 12 * MB);           // 8 MB (4 slices)
    float*    rsum = (float*)   (ws + 20 * MB);           // 128 KB

    pack_kernel<<<(NN / 256) * NN, 256, 0, stream>>>(adj, mask);   // 262144 blocks
    wh_kernel<<<NN, FOUT, 0, stream>>>(h, W, Wh, WhT);
    feat_kernel<<<NN / 256, 256, 0, stream>>>(Wh, a, fsrc, fdst);
    dmax_kernel<<<1, 256, 0, stream>>>(fdst, dmax);

    dim3 agrid(NN / 64, KSPLIT);                                   // 128 x 4
    attn_kernel<<<agrid, 128, 0, stream>>>(mask, WhT, fsrc, fdst, dmax, accW, rsum);

    finalize_kernel<<<(NN * FOUT) / 256, 256, 0, stream>>>(accW, rsum, out);
}
